// NeuralDynamics_69483980914773
// MI455X (gfx1250) — matrix operand order repaired and checked by reading, not compiled, not run on hardware
//
#include <hip/hip_runtime.h>

// ---------------------------------------------------------------------------
// Fused NeuralODE pipeline for gfx1250 (MI455X), bf16 WMMA everywhere.
//   h = tanh(x@w1+b1)@w2+b2 ; RK4-integrate dy/dt = relu(y@wt+b) ; out = y@wo+bo
// One workgroup = 64 nodes (4 waves x 16-row bands). y lives in VGPR C-frags,
// staged to LDS as bf16 to feed WMMA A-frags. wtT (ODE weight) is copied to
// LDS once per WG so the hot RHS-eval B-frags come from ds_load_b128.
// Cold GEMM loops are rolled to bound the load window and stay <256 VGPRs.
// ---------------------------------------------------------------------------

typedef __attribute__((ext_vector_type(16))) __bf16          v16bf;
typedef __attribute__((ext_vector_type(8)))  __bf16          v8bf;
typedef __attribute__((ext_vector_type(8)))  float           v8f;
typedef __attribute__((ext_vector_type(8)))  unsigned short  v8us;

#define DEV static __device__ __forceinline__

DEV unsigned short f2bf_bits(float f) {
    union { float f; unsigned u; } c; c.f = f;
    unsigned u = c.u;
    u += 0x7fffu + ((u >> 16) & 1u);   // round to nearest even
    return (unsigned short)(u >> 16);
}

DEV v8f splat8(float v) {
    v8f r;
    #pragma unroll
    for (int i = 0; i < 8; ++i) r[i] = v;
    return r;
}

// A-fragment (16x32 bf16, M x K) from row-major bf16 tile (LDS).
// Lanes 0-15: row = base+nl, K = kb*32 + {0..7, 16..23}
// Lanes16-31: row = base+nl, K = kb*32 + {8..15, 24..31}
DEV v16bf loadA(const unsigned short* buf, int rowStride, int row, int kb, int hlf) {
    const __bf16* p = (const __bf16*)buf + (size_t)row * rowStride + kb * 32 + hlf * 8;
    v8bf lo = *(const v8bf*)p;
    v8bf hi = *(const v8bf*)(p + 16);
    return __builtin_shufflevector(lo, hi, 0,1,2,3,4,5,6,7,8,9,10,11,12,13,14,15);
}

// 8 consecutive f32 -> v8bf (for building A-frags straight from global x)
DEV v8bf cvt8(const float* p) {
    float4 a = *(const float4*)p;
    float4 b = *(const float4*)(p + 4);
    v8us u;
    u[0] = f2bf_bits(a.x); u[1] = f2bf_bits(a.y);
    u[2] = f2bf_bits(a.z); u[3] = f2bf_bits(a.w);
    u[4] = f2bf_bits(b.x); u[5] = f2bf_bits(b.y);
    u[6] = f2bf_bits(b.z); u[7] = f2bf_bits(b.w);
    union { v8us u; v8bf b; } cc; cc.u = u;
    return cc.b;
}

// A-fragment built directly from global f32 x (row-major, 256 cols)
DEV v16bf loadA_gx(const float* x, long long row, int kb, int hlf) {
    const float* p = x + row * 256 + kb * 32 + hlf * 8;
    v8bf lo = cvt8(p);
    v8bf hi = cvt8(p + 16);
    return __builtin_shufflevector(lo, hi, 0,1,2,3,4,5,6,7,8,9,10,11,12,13,14,15);
}

// B-fragment (32x16 bf16, K x N) from transposed weights WT[N][K] (row-major,
// global OR LDS). Lane n<16: column nrow, K = kb*32+{0..7,16..23}; lane>=16: {8..15,24..31}.
// Two 16-byte loads per lane.
DEV v16bf loadBfr(const unsigned short* wT, int rowStride, int nrow, int kb, int hlf) {
    const __bf16* p = (const __bf16*)wT + (size_t)nrow * rowStride + kb * 32 + hlf * 8;
    v8bf lo = *(const v8bf*)p;
    v8bf hi = *(const v8bf*)(p + 16);
    return __builtin_shufflevector(lo, hi, 0,1,2,3,4,5,6,7,8,9,10,11,12,13,14,15);
}

DEV v8f wmma_bf16(v16bf a, v16bf b, v8f c) {
    return __builtin_amdgcn_wmma_f32_16x16x32_bf16(false, a, false, b, (short)0, c, false, false);
}

// ---------------------------------------------------------------------------
// Prep: dst[n*K + k] = bf16(src[k*Ncols + n])   (transpose + f32->bf16)
// ---------------------------------------------------------------------------
__global__ void node_ode_prepT(const float* __restrict__ src,
                               unsigned short* __restrict__ dst,
                               int K, int Ncols) {
    int i = blockIdx.x * blockDim.x + threadIdx.x;
    if (i >= K * Ncols) return;
    int n = i / K, k = i - n * K;
    dst[i] = f2bf_bits(src[(size_t)k * Ncols + n]);
}

// ---------------------------------------------------------------------------
// Main fused kernel. block = 128 threads (4 wave32), tile = 64 nodes.
// ---------------------------------------------------------------------------
#define SUBSTEPS 2

__global__ __launch_bounds__(128)
void node_ode_main(const float* __restrict__ vt, const float* __restrict__ x,
                   const float* __restrict__ b1, const float* __restrict__ b2,
                   const float* __restrict__ wtb, const float* __restrict__ bo,
                   const unsigned short* __restrict__ w1T,
                   const unsigned short* __restrict__ w2T,
                   const unsigned short* __restrict__ wtT,
                   const unsigned short* __restrict__ woT,
                   float* __restrict__ out, int Nnodes, int T) {
    // LDS: [0,32K)   RK4 acc f32 [64][128]           (band-local per wave)
    //      [32K,64K) wtT bf16 [128][128]             (shared, read-only)
    //      [64K,80K) arg bf16 [64][128]              (band-local per wave)
    __shared__ __align__(16) unsigned char ldsraw[80 * 1024];
    float*          accb = (float*)ldsraw;
    unsigned short* wtTl = (unsigned short*)(ldsraw + 32 * 1024);
    unsigned short* argb = (unsigned short*)(ldsraw + 64 * 1024);

    const int tid  = threadIdx.x;
    const int lane = tid & 31, wv = tid >> 5;
    const int nl = lane & 15, hlf = lane >> 4;
    const int mBase = wv * 16;
    const long long tileBase = (long long)blockIdx.x * 64;

    // ---- stage wtT into LDS (one-time 32KB copy, uint4 granularity) ----
    {
        const uint4* src = (const uint4*)wtT;
        uint4*       dst = (uint4*)wtTl;
        for (int i = tid; i < 2048; i += 128) dst[i] = src[i];
    }
    __syncthreads();   // only barrier: everything below is band-local per wave

    // clamped A row for this lane (OOB rows read row Nnodes-1; their outputs
    // are never stored, and clamped data stays finite so no NaN pollution)
    long long arow = tileBase + mBase + nl;
    if (arow >= Nnodes) arow = Nnodes - 1;

    v8f y[8];

    // ---- GEMM1: U = tanh(X @ w1 + b1); rolled kb loop, one A-frag live ----
    {
        v8f c[8];
        #pragma unroll
        for (int nt = 0; nt < 8; ++nt) c[nt] = splat8(b1[nt * 16 + nl]);
        #pragma unroll 1
        for (int kb = 0; kb < 8; ++kb) {
            v16bf a = loadA_gx(x, arow, kb, hlf);
            #pragma unroll
            for (int nt = 0; nt < 8; ++nt)
                c[nt] = wmma_bf16(a, loadBfr(w1T, 256, nt * 16 + nl, kb, hlf), c[nt]);
        }
        #pragma unroll
        for (int nt = 0; nt < 8; ++nt) {
            #pragma unroll
            for (int r = 0; r < 8; ++r)
                argb[(mBase + r + hlf * 8) * 128 + nt * 16 + nl] = f2bf_bits(tanhf(c[nt][r]));
        }
    }

    // ---- GEMM2: y = U @ w2 + b2; rolled kb loop ----
    {
        #pragma unroll
        for (int nt = 0; nt < 8; ++nt) y[nt] = splat8(b2[nt * 16 + nl]);
        #pragma unroll 1
        for (int kb = 0; kb < 4; ++kb) {
            v16bf a = loadA(argb, 128, mBase + nl, kb, hlf);
            #pragma unroll
            for (int nt = 0; nt < 8; ++nt)
                y[nt] = wmma_bf16(a, loadBfr(w2T, 128, nt * 16 + nl, kb, hlf), y[nt]);
        }
    }

    // keep argb == bf16(y) invariant
    #pragma unroll
    for (int nt = 0; nt < 8; ++nt) {
        #pragma unroll
        for (int r = 0; r < 8; ++r)
            argb[(mBase + r + hlf * 8) * 128 + nt * 16 + nl] = f2bf_bits(y[nt][r]);
    }

    // ---- output GEMM: out[t] = y @ wo + bo (16 N-tiles, partially rolled) ----
    auto outGemm = [&](int t) {
        v16bf afr[4];
        #pragma unroll
        for (int kb = 0; kb < 4; ++kb) afr[kb] = loadA(argb, 128, mBase + nl, kb, hlf);
        #pragma unroll 4
        for (int nt = 0; nt < 16; ++nt) {
            v8f c = splat8(bo[nt * 16 + nl]);
            #pragma unroll
            for (int kb = 0; kb < 4; ++kb)
                c = wmma_bf16(afr[kb], loadBfr(woT, 128, nt * 16 + nl, kb, hlf), c);
            #pragma unroll
            for (int r = 0; r < 8; ++r) {
                long long g = tileBase + mBase + r + hlf * 8;
                if (g < Nnodes)
                    out[((long long)t * Nnodes + g) * 256 + nt * 16 + nl] = c[r];
            }
        }
    };

    // ---- RHS eval: k = relu(arg @ wt + wtb); fully unrolled, wt from LDS ----
    auto feval = [&](v8f (&k)[8]) {
        #pragma unroll
        for (int nt = 0; nt < 8; ++nt) k[nt] = splat8(wtb[nt * 16 + nl]);
        #pragma unroll
        for (int kb = 0; kb < 4; ++kb) {
            v16bf a = loadA(argb, 128, mBase + nl, kb, hlf);
            #pragma unroll
            for (int nt = 0; nt < 8; ++nt)
                k[nt] = wmma_bf16(a, loadBfr(wtTl, 128, nt * 16 + nl, kb, hlf), k[nt]);
        }
        #pragma unroll
        for (int nt = 0; nt < 8; ++nt) {
            #pragma unroll
            for (int r = 0; r < 8; ++r) k[nt][r] = fmaxf(k[nt][r], 0.f);
        }
    };

    outGemm(0);

    // ---- classic RK4, SUBSTEPS per output interval ----
    for (int ti = 0; ti + 1 < T; ++ti) {
        float dt = (vt[ti + 1] - vt[ti]) * (1.0f / SUBSTEPS);
        for (int s = 0; s < SUBSTEPS; ++s) {
            v8f k[8];
            feval(k);                                   // k1
            #pragma unroll
            for (int nt = 0; nt < 8; ++nt) {
                #pragma unroll
                for (int r = 0; r < 8; ++r) {
                    int idx = (mBase + r + hlf * 8) * 128 + nt * 16 + nl;
                    float kv = k[nt][r];
                    accb[idx] = kv;
                    argb[idx] = f2bf_bits(y[nt][r] + 0.5f * dt * kv);
                }
            }
            feval(k);                                   // k2
            #pragma unroll
            for (int nt = 0; nt < 8; ++nt) {
                #pragma unroll
                for (int r = 0; r < 8; ++r) {
                    int idx = (mBase + r + hlf * 8) * 128 + nt * 16 + nl;
                    float kv = k[nt][r];
                    accb[idx] += 2.f * kv;
                    argb[idx] = f2bf_bits(y[nt][r] + 0.5f * dt * kv);
                }
            }
            feval(k);                                   // k3
            #pragma unroll
            for (int nt = 0; nt < 8; ++nt) {
                #pragma unroll
                for (int r = 0; r < 8; ++r) {
                    int idx = (mBase + r + hlf * 8) * 128 + nt * 16 + nl;
                    float kv = k[nt][r];
                    accb[idx] += 2.f * kv;
                    argb[idx] = f2bf_bits(y[nt][r] + dt * kv);
                }
            }
            feval(k);                                   // k4 + combine
            #pragma unroll
            for (int nt = 0; nt < 8; ++nt) {
                #pragma unroll
                for (int r = 0; r < 8; ++r) {
                    int idx = (mBase + r + hlf * 8) * 128 + nt * 16 + nl;
                    float yn = y[nt][r] + (dt * (1.f / 6.f)) * (accb[idx] + k[nt][r]);
                    y[nt][r] = yn;
                    argb[idx] = f2bf_bits(yn);
                }
            }
        }
        outGemm(ti + 1);
    }
}

// ---------------------------------------------------------------------------
extern "C" void kernel_launch(void* const* d_in, const int* in_sizes, int n_in,
                              void* d_out, int out_size, void* d_ws, size_t ws_size,
                              hipStream_t stream) {
    const float* vt  = (const float*)d_in[0];
    const float* x   = (const float*)d_in[1];
    const float* w1  = (const float*)d_in[2];
    const float* b1  = (const float*)d_in[3];
    const float* w2  = (const float*)d_in[4];
    const float* b2  = (const float*)d_in[5];
    const float* wt  = (const float*)d_in[6];
    const float* wtb = (const float*)d_in[7];
    const float* wo  = (const float*)d_in[8];
    const float* bo  = (const float*)d_in[9];
    float* out = (float*)d_out;

    const int T = in_sizes[0];
    const int N = in_sizes[1] / 256;

    unsigned short* ws  = (unsigned short*)d_ws;
    unsigned short* w1T = ws;                         // [128][256] bf16
    unsigned short* w2T = ws + 32768;                 // [128][128]
    unsigned short* wtT = ws + 32768 + 16384;         // [128][128]
    unsigned short* woT = ws + 32768 + 16384 + 16384; // [256][128]

    node_ode_prepT<<<(32768 + 255) / 256, 256, 0, stream>>>(w1, w1T, 256, 128);
    node_ode_prepT<<<(16384 + 255) / 256, 256, 0, stream>>>(w2, w2T, 128, 128);
    node_ode_prepT<<<(16384 + 255) / 256, 256, 0, stream>>>(wt, wtT, 128, 128);
    node_ode_prepT<<<(32768 + 255) / 256, 256, 0, stream>>>(wo, woT, 128, 256);

    node_ode_main<<<(N + 63) / 64, 128, 0, stream>>>(
        vt, x, b1, b2, wtb, bo, w1T, w2T, wtT, woT, out, N, T);
}
